// ToyAttnSublayer_20074677141602
// MI455X (gfx1250) — matrix operand order repaired and checked by reading, not compiled, not run on hardware
//
#include <hip/hip_runtime.h>

typedef __bf16 bf16;
typedef __attribute__((ext_vector_type(16))) __bf16 v16bf;
typedef __attribute__((ext_vector_type(8)))  __bf16 v8bf;
typedef __attribute__((ext_vector_type(8)))  float  v8f;

#define TSEQ 8192
#define DMODEL 512

__device__ inline bf16 f2bf(float f) {
  union { float f; unsigned u; } a; a.f = f;
  unsigned r = a.u + 0x7FFFu + ((a.u >> 16) & 1u);   // round-to-nearest-even
  union { unsigned short s; bf16 b; } o; o.s = (unsigned short)(r >> 16);
  return o.b;
}

__device__ inline v8f wmma_bf16(v16bf a, v16bf b, v8f c) {
  return __builtin_amdgcn_wmma_f32_16x16x32_bf16(false, a, false, b, (short)0, c,
                                                 false, false);
}

// ---- Fragment loaders (A/B in contiguous k order, same for both) -----------------------

// A fragment: 16x32 (MxK) bf16 from row-major [*, ld].
// lanes 0-15: row=lane, K = k0..k0+15; lanes 16-31: K = k0+16..k0+31 (contiguous).
__device__ inline v16bf load_A16(const bf16* base, int ld, int row0, int k0) {
  const int lane = threadIdx.x & 31;
  const int r    = lane & 15;
  const int kh   = (lane >> 4) << 4;                 // 0 or 16
  const bf16* p  = base + (size_t)(row0 + r) * ld + (k0 + kh);
  v8bf lo = *(const v8bf*)p;
  v8bf hi = *(const v8bf*)(p + 8);
  return __builtin_shufflevector(lo, hi, 0,1,2,3,4,5,6,7,8,9,10,11,12,13,14,15);
}

// B fragment (32x16 KxN) where B[k][n] = S[n][k], S row-major [N, ld=K].
// lanes 0-15: col=lane, K = k0..k0+15; lanes 16-31: K = k0+16..k0+31 (contiguous).
__device__ inline v16bf load_B16_NT(const bf16* base, int ld, int n0, int k0) {
  const int lane = threadIdx.x & 31;
  const int n    = lane & 15;
  const int kh   = (lane >> 4) << 4;                 // 0 or 16
  const bf16* p  = base + (size_t)(n0 + n) * ld + (k0 + kh);
  return *(const v16bf*)p;                           // 32B contiguous
}

// C (16x16 f32) store: VGPR r -> row r (lanes 0-15) / row r+8 (lanes 16-31).
__device__ inline void store_C_bf16(bf16* dst, int ld, int row0, int col0, v8f c) {
  const int lane = threadIdx.x & 31;
  const int n    = lane & 15;
  const int mb   = (lane >> 4) << 3;
#pragma unroll
  for (int r = 0; r < 8; ++r)
    dst[(size_t)(row0 + mb + r) * ld + col0 + n] = f2bf(c[r]);
}

// Transposed C store: dst[col][row], dst row-major [N, ldT].
__device__ inline void store_C_bf16_T(bf16* dst, int ldT, int row0, int col0, v8f c) {
  const int lane = threadIdx.x & 31;
  const int n    = lane & 15;
  const int mb   = (lane >> 4) << 3;
#pragma unroll
  for (int r = 0; r < 8; ++r)
    dst[(size_t)(col0 + n) * ldT + row0 + mb + r] = f2bf(c[r]);
}

// ---- Kernel 1: LayerNorm (wave per row) ------------------------------------
__global__ __launch_bounds__(256)
void ln_kernel(const float* __restrict__ h, const float* __restrict__ w,
               const float* __restrict__ b, bf16* __restrict__ hn) {
  const int wave = (blockIdx.x * blockDim.x + threadIdx.x) >> 5;
  const int lane = threadIdx.x & 31;
  if (wave >= TSEQ) return;
  const float4* row = (const float4*)(h + (size_t)wave * DMODEL);
  float4 x[4];
  float s = 0.f, ss = 0.f;
#pragma unroll
  for (int i = 0; i < 4; ++i) {
    x[i] = row[i * 32 + lane];
    s  += x[i].x + x[i].y + x[i].z + x[i].w;
    ss += x[i].x*x[i].x + x[i].y*x[i].y + x[i].z*x[i].z + x[i].w*x[i].w;
  }
#pragma unroll
  for (int m = 16; m >= 1; m >>= 1) {
    s  += __shfl_xor(s, m);
    ss += __shfl_xor(ss, m);
  }
  const float mu = s * (1.0f / DMODEL);
  const float var = ss * (1.0f / DMODEL) - mu * mu;
  const float rs = rsqrtf(var + 1e-5f);
  bf16* orow = hn + (size_t)wave * DMODEL;
#pragma unroll
  for (int i = 0; i < 4; ++i) {
    const int c = (i * 32 + lane) * 4;
    float v0 = (x[i].x - mu) * rs * w[c + 0] + b[c + 0];
    float v1 = (x[i].y - mu) * rs * w[c + 1] + b[c + 1];
    float v2 = (x[i].z - mu) * rs * w[c + 2] + b[c + 2];
    float v3 = (x[i].w - mu) * rs * w[c + 3] + b[c + 3];
    orow[c + 0] = f2bf(v0); orow[c + 1] = f2bf(v1);
    orow[c + 2] = f2bf(v2); orow[c + 3] = f2bf(v3);
  }
}

// ---- Kernel 2: f32 -> bf16 conversion --------------------------------------
__global__ __launch_bounds__(256)
void cvt_kernel(const float* __restrict__ src, bf16* __restrict__ dst, int n) {
  int i = blockIdx.x * blockDim.x + threadIdx.x;
  if (i < n) dst[i] = f2bf(src[i]);
}

// ---- Kernel 3: QKV GEMM  C = hn @ W.T (one wave -> 16x64 tile) -------------
// z==2 (V) stores its result TRANSPOSED: Vt[feature][t], ld = TSEQ, so the
// attention P.V B-fragments become contiguous 32B loads.
__global__ __launch_bounds__(32)
void qkv_gemm(const bf16* __restrict__ hn,
              const bf16* __restrict__ Wq, const bf16* __restrict__ Wk,
              const bf16* __restrict__ Wv,
              bf16* __restrict__ Q, bf16* __restrict__ K, bf16* __restrict__ Vt) {
  const bf16* W;
  if (blockIdx.z == 0)      W = Wq;
  else if (blockIdx.z == 1) W = Wk;
  else                      W = Wv;
  const int m0 = blockIdx.x * 16;
  const int n0 = blockIdx.y * 64;
  v8f acc[4] = {};
#pragma unroll
  for (int kk = 0; kk < 16; ++kk) {
    v16bf a = load_A16(hn, DMODEL, m0, kk * 32);
#pragma unroll
    for (int t = 0; t < 4; ++t) {
      v16bf bb = load_B16_NT(W, DMODEL, n0 + 16 * t, kk * 32);
      acc[t] = wmma_bf16(a, bb, acc[t]);
    }
  }
  if (blockIdx.z == 0) {
#pragma unroll
    for (int t = 0; t < 4; ++t) store_C_bf16(Q, DMODEL, m0, n0 + 16 * t, acc[t]);
  } else if (blockIdx.z == 1) {
#pragma unroll
    for (int t = 0; t < 4; ++t) store_C_bf16(K, DMODEL, m0, n0 + 16 * t, acc[t]);
  } else {
#pragma unroll
    for (int t = 0; t < 4; ++t) store_C_bf16_T(Vt, TSEQ, m0, n0 + 16 * t, acc[t]);
  }
}

// ---- Kernel 4: causal flash attention (one wave per 16-row query tile) -----
__global__ __launch_bounds__(32)
void attn_kernel(const bf16* __restrict__ Q, const bf16* __restrict__ Km,
                 const bf16* __restrict__ Vt, bf16* __restrict__ O) {
  __shared__ bf16 Qs[16 * DMODEL];               // 16KB: query tile staging
  __shared__ bf16 Pbuf[16 * 32];                 // 1KB: P tile C->A relayout
  const int m0    = blockIdx.x * 16;
  const int lane  = threadIdx.x & 31;
  const int rbase = (lane >> 4) << 3;            // stats: VGPR r -> row r+rbase
  const float sc  = 0.044194173824159216f;       // 1/sqrt(512)

  // stage Q tile (16 rows, contiguous) into LDS, 16B vector copies
  {
    const v8bf* src = (const v8bf*)(Q + (size_t)m0 * DMODEL);
    v8bf* dst = (v8bf*)Qs;
#pragma unroll
    for (int i = 0; i < 32; ++i) dst[i * 32 + lane] = src[i * 32 + lane];
  }
  __syncthreads();

  v8f outacc[32] = {};
  float m[8], l[8];
#pragma unroll
  for (int r = 0; r < 8; ++r) { m[r] = -1e30f; l[r] = 0.f; }

  const int nb = (m0 + 47) >> 5;                 // ceil((m0+16)/32) key blocks
  for (int jb = 0; jb < nb; ++jb) {
    const int key0 = jb * 32;
    // scores: S[16x32] = Q_tile . K_block^T, two 16x16 C tiles
    v8f s0 = {}, s1 = {};
#pragma unroll
    for (int kk = 0; kk < 16; ++kk) {
      v16bf a  = load_A16(Qs, DMODEL, 0, kk * 32);          // from LDS
      v16bf b0 = load_B16_NT(Km, DMODEL, key0,      kk * 32);
      v16bf b1 = load_B16_NT(Km, DMODEL, key0 + 16, kk * 32);
      s0 = wmma_bf16(a, b0, s0);
      s1 = wmma_bf16(a, b1, s1);
    }
    // mask + online softmax (rows live in VGPR index + lane-half)
    const int coln = key0 + (lane & 15);
    float scl[8];
#pragma unroll
    for (int r = 0; r < 8; ++r) {
      const int row = m0 + r + rbase;
      float a0 = s0[r] * sc;
      float a1 = s1[r] * sc;
      if (coln > row)      a0 = -1e30f;
      if (coln + 16 > row) a1 = -1e30f;
      float rm = fmaxf(a0, a1);
      rm = fmaxf(rm, __shfl_xor(rm, 1));
      rm = fmaxf(rm, __shfl_xor(rm, 2));
      rm = fmaxf(rm, __shfl_xor(rm, 4));
      rm = fmaxf(rm, __shfl_xor(rm, 8));
      const float mn = fmaxf(m[r], rm);
      const float p0 = __expf(a0 - mn);
      const float p1 = __expf(a1 - mn);
      float rsum = p0 + p1;
      rsum += __shfl_xor(rsum, 1);
      rsum += __shfl_xor(rsum, 2);
      rsum += __shfl_xor(rsum, 4);
      rsum += __shfl_xor(rsum, 8);
      scl[r] = __expf(m[r] - mn);
      l[r] = l[r] * scl[r] + rsum;
      m[r] = mn;
      const int prow = r + rbase;
      const int pc   = lane & 15;
      Pbuf[prow * 32 + pc]      = f2bf(p0);
      Pbuf[prow * 32 + pc + 16] = f2bf(p1);
    }
    __syncthreads();                             // single wave: lowers to ds waits
    v16bf pf = load_A16(Pbuf, 32, 0, 0);
    __syncthreads();
    // out = out*scale + P . V_block  (Vt contiguous B-frags, 32 col tiles)
#pragma unroll
    for (int n = 0; n < 32; ++n) {
      v16bf vf = load_B16_NT(Vt, TSEQ, n * 16, key0);
      v8f o = outacc[n];
#pragma unroll
      for (int r = 0; r < 8; ++r) o[r] *= scl[r];
      outacc[n] = wmma_bf16(pf, vf, o);
    }
  }
  // normalize and store
  float inv[8];
#pragma unroll
  for (int r = 0; r < 8; ++r) inv[r] = 1.0f / l[r];
#pragma unroll
  for (int n = 0; n < 32; ++n) {
    v8f o = outacc[n];
#pragma unroll
    for (int r = 0; r < 8; ++r) o[r] *= inv[r];
    store_C_bf16(O, DMODEL, m0, n * 16, o);
  }
}

// ---- Kernel 5: output projection + residual (f32 out) ----------------------
__global__ __launch_bounds__(32)
void proj_resid(const bf16* __restrict__ AO, const bf16* __restrict__ Wo,
                const float* __restrict__ h, float* __restrict__ out) {
  const int m0 = blockIdx.x * 16;
  const int n0 = blockIdx.y * 64;
  v8f acc[4] = {};
#pragma unroll
  for (int kk = 0; kk < 16; ++kk) {
    v16bf a = load_A16(AO, DMODEL, m0, kk * 32);
#pragma unroll
    for (int t = 0; t < 4; ++t) {
      v16bf bb = load_B16_NT(Wo, DMODEL, n0 + 16 * t, kk * 32);
      acc[t] = wmma_bf16(a, bb, acc[t]);
    }
  }
  const int lane = threadIdx.x & 31;
  const int n    = lane & 15;
  const int mb   = (lane >> 4) << 3;
#pragma unroll
  for (int t = 0; t < 4; ++t) {
#pragma unroll
    for (int r = 0; r < 8; ++r) {
      size_t idx = (size_t)(m0 + mb + r) * DMODEL + n0 + 16 * t + n;
      out[idx] = h[idx] + acc[t][r];
    }
  }
}

extern "C" void kernel_launch(void* const* d_in, const int* in_sizes, int n_in,
                              void* d_out, int out_size, void* d_ws, size_t ws_size,
                              hipStream_t stream) {
  const float* h    = (const float*)d_in[0];
  const float* ln_w = (const float*)d_in[1];
  const float* ln_b = (const float*)d_in[2];
  const float* w_q  = (const float*)d_in[3];
  const float* w_k  = (const float*)d_in[4];
  const float* w_v  = (const float*)d_in[5];
  const float* w_o  = (const float*)d_in[6];
  float* out = (float*)d_out;

  const size_t TD = (size_t)TSEQ * DMODEL;   // 4 Mi elements
  const size_t DD = (size_t)DMODEL * DMODEL; // 256 Ki elements
  char* ws = (char*)d_ws;
  bf16* hn  = (bf16*)ws;                 ws += TD * sizeof(bf16);
  bf16* Qb  = (bf16*)ws;                 ws += TD * sizeof(bf16);
  bf16* Kb  = (bf16*)ws;                 ws += TD * sizeof(bf16);
  bf16* Vtb = (bf16*)ws;                 ws += TD * sizeof(bf16);  // [DMODEL][TSEQ]
  bf16* AOb = (bf16*)ws;                 ws += TD * sizeof(bf16);
  bf16* wqb = (bf16*)ws;                 ws += DD * sizeof(bf16);
  bf16* wkb = (bf16*)ws;                 ws += DD * sizeof(bf16);
  bf16* wvb = (bf16*)ws;                 ws += DD * sizeof(bf16);
  bf16* wob = (bf16*)ws;                 ws += DD * sizeof(bf16);

  const int cvtBlocks = (int)(DD / 256);
  cvt_kernel<<<cvtBlocks, 256, 0, stream>>>(w_q, wqb, (int)DD);
  cvt_kernel<<<cvtBlocks, 256, 0, stream>>>(w_k, wkb, (int)DD);
  cvt_kernel<<<cvtBlocks, 256, 0, stream>>>(w_v, wvb, (int)DD);
  cvt_kernel<<<cvtBlocks, 256, 0, stream>>>(w_o, wob, (int)DD);

  ln_kernel<<<TSEQ / 8, 256, 0, stream>>>(h, ln_w, ln_b, hn);

  qkv_gemm<<<dim3(TSEQ / 16, DMODEL / 64, 3), 32, 0, stream>>>(
      hn, wqb, wkb, wvb, Qb, Kb, Vtb);

  attn_kernel<<<TSEQ / 16, 32, 0, stream>>>(Qb, Kb, Vtb, AOb);

  proj_resid<<<dim3(TSEQ / 16, DMODEL / 64), 32, 0, stream>>>(AOb, wob, h, out);
}
